// GraphEva_38362647887874
// MI455X (gfx1250) — matrix operand order repaired and checked by reading, not compiled, not run on hardware
//
#include <hip/hip_runtime.h>
#include <hip/hip_bf16.h>
#include <math.h>

#define TPB 256

typedef __bf16 bf16_t;
typedef __attribute__((ext_vector_type(16))) __bf16 v16bf;
typedef __attribute__((ext_vector_type(8)))  __bf16 v8bf;
typedef __attribute__((ext_vector_type(8)))  float  v8f;

static const int kEXER = 10000;
static const int kSTU  = 20000;
static const int kN    = 30000;   // kEXER + kSTU
static const int kD    = 128;
static const int kH    = 3;
static const int kHD   = 384;     // H*D
static const int kBATCH= 2048;

// fenc(-INFINITY): bits(-inf)=0xFF800000 -> ~bits
#define ENC_NEG_INF 0x007FFFFFu

// ---------------- device helpers ----------------

__device__ __forceinline__ unsigned fenc(float f) {
  unsigned u = __float_as_uint(f);
  return (u & 0x80000000u) ? ~u : (u | 0x80000000u);
}
__device__ __forceinline__ float fdec(unsigned u) {
  return __uint_as_float((u & 0x80000000u) ? (u & 0x7FFFFFFFu) : ~u);
}
__device__ __forceinline__ float wred(float v) {
#pragma unroll
  for (int o = 16; o > 0; o >>= 1) v += __shfl_xor(v, o, 32);
  return v;
}

// ---------------- generic kernels ----------------

__global__ void k_fill_f32(float* __restrict__ p, float v, long long n) {
  long long i = (long long)blockIdx.x * blockDim.x + threadIdx.x;
  if (i < n) p[i] = v;
}
__global__ void k_fill_u32(unsigned* __restrict__ p, unsigned v, long long n) {
  long long i = (long long)blockIdx.x * blockDim.x + threadIdx.x;
  if (i < n) p[i] = v;
}
__global__ void k_cvt_bf16(const float* __restrict__ s, bf16_t* __restrict__ d, long long n) {
  long long i = (long long)blockIdx.x * blockDim.x + threadIdx.x;
  if (i < n) d[i] = (bf16_t)s[i];
}
__global__ void k_add(const float* __restrict__ x, const float* __restrict__ y,
                      float* __restrict__ o, long long n) {
  long long i = (long long)blockIdx.x * blockDim.x + threadIdx.x;
  if (i < n) o[i] = x[i] + y[i];
}

// Pack a row-major f32 weight matrix B[K,Nc] into the CDNA5 WMMA B-fragment
// order (bf16): Bp[((nt*(K/32)+kb)*32 + lane)*16 + i], so each lane's 16
// values for one (n-tile, k-block) are contiguous 32 bytes.
__global__ void k_packW(const float* __restrict__ B, bf16_t* __restrict__ Bp,
                        int K, int Nc, long long total) {
  long long t = (long long)blockIdx.x * blockDim.x + threadIdx.x;
  if (t >= total) return;
  int i    = (int)(t & 15);
  int lane = (int)((t >> 4) & 31);
  int kblocks = K >> 5;
  long long r = t >> 9;                 // nt*kblocks + kb
  int kb = (int)(r % kblocks);
  int nt = (int)(r / kblocks);
  int n  = lane & 15;
  int hi = lane >> 4;
  int g = i >> 1, s2 = i & 1;
  int k = (kb << 5) + ((i >> 3) << 4) + (hi << 3) + (i & 7);
  Bp[t] = (bf16_t)B[(size_t)k * Nc + nt * 16 + n];
}

// ---------------- WMMA GEMM: C[M,Nc] = A[M,K]@B[K,Nc] (+bias) ----------------
// A row-major bf16; B pre-packed fragment-order bf16; C f32.
// One wave computes a 16x64 strip (4 N-tiles): A fragment loaded once per
// 32-wide K step and reused by 4 v_wmma_f32_16x16x32_bf16. Next k-block's
// A/B lines are prefetched branch-free (clamped index, WGP-near locality)
// while the current WMMAs run.

__global__ __launch_bounds__(TPB)
void k_gemm_bf16(const bf16_t* __restrict__ A, const bf16_t* __restrict__ Bp,
                 const float* __restrict__ bias, float* __restrict__ C,
                 int M, int K, int Nc) {
  int tilesN4 = Nc >> 6;
  long long gid = (long long)blockIdx.x * blockDim.x + threadIdx.x;
  int wave = (int)(gid >> 5);
  int lane = threadIdx.x & 31;
  int tot = (M >> 4) * tilesN4;
  if (wave >= tot) return;
  int tm  = (wave / tilesN4) << 4;
  int tn0 = (wave % tilesN4) << 6;
  int m  = lane & 15;
  int hi = lane >> 4;
  v8f acc[4];
#pragma unroll
  for (int t = 0; t < 4; ++t) acc[t] = (v8f){0.f,0.f,0.f,0.f,0.f,0.f,0.f,0.f};
  const bf16_t* Ar = A + (size_t)(tm + m) * K;
  int kblocks = K >> 5;
  int kbA = hi ? 8 : 0;
  int nt0 = tn0 >> 4;
  for (int kb = 0; kb < kblocks; ++kb) {
    int k0 = kb << 5;
    v8bf a0 = *(const v8bf*)(Ar + k0 + kbA);        // K = kbA..kbA+7
    v8bf a1 = *(const v8bf*)(Ar + k0 + 16 + kbA);   // K = 16+kbA..23+kbA
    v16bf av;
#pragma unroll
    for (int i = 0; i < 8; ++i) { av[i] = a0[i]; av[8 + i] = a1[i]; }
    // warm next k-block while this one computes (branch-free, in-bounds)
    int kbn = (kb + 1 < kblocks) ? (kb + 1) : kb;
    __builtin_prefetch(Ar + (kbn << 5) + kbA, 0, 3);
    __builtin_prefetch(Bp + (((size_t)nt0 * kblocks + kbn) << 9) + (lane << 4), 0, 3);
#pragma unroll
    for (int t = 0; t < 4; ++t) {
      int nt = nt0 + t;
      const bf16_t* bp = Bp + (((size_t)nt * kblocks + kb) << 9) + (lane << 4);
      v16bf bv = *(const v16bf*)bp;
      acc[t] = __builtin_amdgcn_wmma_f32_16x16x32_bf16(false, av, false, bv,
                                                       (short)0, acc[t], false, false);
    }
  }
  int r0 = tm + (hi ? 8 : 0);
#pragma unroll
  for (int t = 0; t < 4; ++t) {
    int tn = tn0 + (t << 4);
    float bc = bias ? bias[tn + m] : 0.f;
#pragma unroll
    for (int g = 0; g < 8; ++g)
      C[(size_t)(r0 + g) * Nc + tn + m] = acc[t][g] + bc;
  }
}

// ---------------- GAT layer kernels ----------------

// el[n,h] = f[n,h,:]·al[h,:]; er likewise. One wave per (n,h).
__global__ void k_el_er(const float* __restrict__ f, const float* __restrict__ al,
                        const float* __restrict__ ar, float* __restrict__ el,
                        float* __restrict__ er) {
  long long gw = ((long long)blockIdx.x * blockDim.x + threadIdx.x) >> 5;
  int lane = threadIdx.x & 31;
  if (gw >= (long long)kN * kH) return;
  int n = (int)(gw / kH), h = (int)(gw % kH);
  const float* fr = f + (size_t)n * kHD + h * kD;
  const float* a0 = al + h * kD;
  const float* a1 = ar + h * kD;
  float sl = 0.f, sr = 0.f;
  for (int d = lane; d < kD; d += 32) {
    float fv = fr[d];
    sl += fv * a0[d];
    sr += fv * a1[d];
  }
  sl = wred(sl); sr = wred(sr);
  if (lane == 0) { el[gw] = sl; er[gw] = sr; }
}

__device__ __forceinline__ float lrelu(float v) { return v > 0.f ? v : 0.2f * v; }

__global__ void k_edge1_max(const float* __restrict__ el, const float* __restrict__ er,
                            const int* __restrict__ src, const int* __restrict__ dst,
                            unsigned* __restrict__ m1, long long neH) {
  long long t = (long long)blockIdx.x * blockDim.x + threadIdx.x;
  if (t >= neH) return;
  int e = (int)(t / kH), h = (int)(t % kH);
  float v = lrelu(el[(size_t)src[e] * kH + h] + er[(size_t)dst[e] * kH + h]);
  atomicMax(&m1[(size_t)dst[e] * kH + h], fenc(v));
}

__global__ void k_edge1_exp(const float* __restrict__ el, const float* __restrict__ er,
                            const int* __restrict__ src, const int* __restrict__ dst,
                            const unsigned* __restrict__ m1, float* __restrict__ s1,
                            float* __restrict__ ex1, long long neH) {
  long long t = (long long)blockIdx.x * blockDim.x + threadIdx.x;
  if (t >= neH) return;
  int e = (int)(t / kH), h = (int)(t % kH);
  float v = lrelu(el[(size_t)src[e] * kH + h] + er[(size_t)dst[e] * kH + h]);
  float ex = expf(v - fdec(m1[(size_t)dst[e] * kH + h]));
  ex1[t] = ex;
  atomicAdd(&s1[(size_t)dst[e] * kH + h], ex);
}

// rst[dst, h, d] += (ex/s) * f[src, h, d]
__global__ void k_agg1(const float* __restrict__ f, const float* __restrict__ ex1,
                       const float* __restrict__ s1, const int* __restrict__ src,
                       const int* __restrict__ dst, float* __restrict__ rst,
                       long long total) {
  long long t = (long long)blockIdx.x * blockDim.x + threadIdx.x;
  if (t >= total) return;
  int e = (int)(t / kHD);
  int r = (int)(t - (long long)e * kHD);
  int h = r >> 7;
  float alpha = ex1[(size_t)e * kH + h] / s1[(size_t)dst[e] * kH + h];
  atomicAdd(&rst[(size_t)dst[e] * kHD + r], alpha * f[(size_t)src[e] * kHD + r]);
}

__global__ void k_hb(const float* __restrict__ rst, const float* __restrict__ gb,
                     bf16_t* __restrict__ hb, long long total) {
  long long i = (long long)blockIdx.x * blockDim.x + threadIdx.x;
  if (i >= total) return;
  int c = (int)(i % kHD);
  hb[i] = (bf16_t)(rst[i] + gb[c]);
}

// zl[n] = z[n,:]·aW[0:128]; zr[n] = z[n,:]·aW[128:256]. One wave per n.
__global__ void k_zlzr(const float* __restrict__ z, const float* __restrict__ aW,
                       float* __restrict__ zl, float* __restrict__ zr) {
  long long gw = ((long long)blockIdx.x * blockDim.x + threadIdx.x) >> 5;
  int lane = threadIdx.x & 31;
  if (gw >= kN) return;
  const float* zr_ = z + gw * kD;
  float sl = 0.f, sr = 0.f;
  for (int d = lane; d < kD; d += 32) {
    float v = zr_[d];
    sl += v * aW[d];
    sr += v * aW[kD + d];
  }
  sl = wred(sl); sr = wred(sr);
  if (lane == 0) { zl[gw] = sl; zr[gw] = sr; }
}

__global__ void k_edge2_max(const float* __restrict__ zl, const float* __restrict__ zr,
                            const int* __restrict__ src, const int* __restrict__ dst,
                            unsigned* __restrict__ m2, long long ne) {
  long long e = (long long)blockIdx.x * blockDim.x + threadIdx.x;
  if (e >= ne) return;
  atomicMax(&m2[dst[e]], fenc(zl[src[e]] + zr[dst[e]]));
}

__global__ void k_edge2_exp(const float* __restrict__ zl, const float* __restrict__ zr,
                            const int* __restrict__ src, const int* __restrict__ dst,
                            const unsigned* __restrict__ m2, float* __restrict__ s2,
                            float* __restrict__ ex2, long long ne) {
  long long e = (long long)blockIdx.x * blockDim.x + threadIdx.x;
  if (e >= ne) return;
  float ex = expf(zl[src[e]] + zr[dst[e]] - fdec(m2[dst[e]]));
  ex2[e] = ex;
  atomicAdd(&s2[dst[e]], ex);
}

__global__ void k_agg2(const float* __restrict__ z, const float* __restrict__ ex2,
                       const float* __restrict__ s2, const int* __restrict__ src,
                       const int* __restrict__ dst, float* __restrict__ out,
                       long long total) {
  long long t = (long long)blockIdx.x * blockDim.x + threadIdx.x;
  if (t >= total) return;
  int e = (int)(t >> 7);
  int d = (int)(t & 127);
  float a = ex2[e] / s2[dst[e]];
  atomicAdd(&out[(size_t)dst[e] * kD + d], a * z[(size_t)src[e] * kD + d]);
}

// ---------------- fusion / SSL kernels ----------------

__global__ void k_cat4(const float* __restrict__ a, const float* __restrict__ b,
                       const float* __restrict__ c, const float* __restrict__ d4,
                       bf16_t* __restrict__ cat, long long total) {
  long long t = (long long)blockIdx.x * blockDim.x + threadIdx.x;
  if (t >= total) return;
  long long n = t >> 9;
  int col = (int)(t & 511);
  int sel = col >> 7, d = col & 127;
  const float* src = (sel == 0) ? a : (sel == 1) ? b : (sel == 2) ? c : d4;
  cat[t] = (bf16_t)src[n * kD + d];
}

__global__ void k_cat2(const float* __restrict__ a, const float* __restrict__ b,
                       bf16_t* __restrict__ cat, long long total) {
  long long t = (long long)blockIdx.x * blockDim.x + threadIdx.x;
  if (t >= total) return;
  long long n = t >> 8;
  int col = (int)(t & 255);
  int sel = col >> 7, d = col & 127;
  const float* src = sel ? b : a;
  cat[t] = (bf16_t)src[n * kD + d];
}

__global__ void k_gather_norm(const float* __restrict__ tab, const int* __restrict__ ids,
                              float* __restrict__ out, int rows) {
  long long gw = ((long long)blockIdx.x * blockDim.x + threadIdx.x) >> 5;
  int lane = threadIdx.x & 31;
  if (gw >= rows) return;
  const float* r = tab + (size_t)ids[gw] * kD;
  float v[4]; float ss = 0.f;
#pragma unroll
  for (int j = 0; j < 4; ++j) { v[j] = r[lane + 32 * j]; ss += v[j] * v[j]; }
  ss = wred(ss);
  float dn = fmaxf(sqrtf(ss), 1e-12f);
#pragma unroll
  for (int j = 0; j < 4; ++j) out[gw * kD + lane + 32 * j] = v[j] / dn;
}

__global__ void k_colsum(const float* __restrict__ p, float* __restrict__ S, long long n) {
  long long t = (long long)blockIdx.x * blockDim.x + threadIdx.x;
  if (t >= n) return;
  atomicAdd(&S[t & 127], p[t]);
}

// loss_i = -log(clip(exp(n_i·p_i/0.2) / (n_i·S/0.2 + 1e-8), 1e-8))
__global__ void k_loss(const float* __restrict__ nv, const float* __restrict__ pv,
                       const float* __restrict__ S, float* __restrict__ loss) {
  long long gw = ((long long)blockIdx.x * blockDim.x + threadIdx.x) >> 5;
  int lane = threadIdx.x & 31;
  if (gw >= kBATCH) return;
  const float* ni = nv + gw * kD;
  const float* pi = pv + gw * kD;
  float dg = 0.f, rs = 0.f;
#pragma unroll
  for (int j = 0; j < 4; ++j) {
    int d = lane + 32 * j;
    float n = ni[d];
    dg += n * pi[d];
    rs += n * S[d];
  }
  dg = wred(dg); rs = wred(rs);
  if (lane == 0) {
    float divided = expf(dg * 5.0f) / (rs * 5.0f + 1e-8f);
    divided = fmaxf(divided, 1e-8f);
    atomicAdd(loss, -logf(divided));
  }
}

// ---------------- host orchestration ----------------

struct Scratch {
  float *x0, *x2, *A, *Bf, *Cf, *Df, *BP, *stu1, *stu1p, *stu2p;
  float *f, *rst, *z, *el, *er, *s1, *zl, *zr, *s2, *ex1, *ex2, *fus, *bn, *pn, *S;
  unsigned *m1u, *m2u;
  bf16_t *xb, *hb, *catb, *gWp, *fWp, *f1Wp, *f2Wp;
};

struct Wts {
  const float *attn_l, *attn_r, *gat_b, *fc_b, *attn_fc_W;
};

static inline int gb_(long long n) { return (int)((n + TPB - 1) / TPB); }
static inline int gemm_blocks(int M, int Nc) {
  int waves = (M >> 4) * (Nc >> 6);
  return (waves + 7) / 8;
}

static void run_layer(hipStream_t s, const Scratch& w, const Wts& p,
                      int li, const int* src, const int* dst, int ne, float* out) {
  long long neH = (long long)ne * kH;
  long long neHD = (long long)ne * kHD;
  long long neD = (long long)ne * kD;
  long long NH = (long long)kN * kH;
  long long NHD = (long long)kN * kHD;
  long long ND = (long long)kN * kD;

  // f = xb @ gat_W[li]
  k_gemm_bf16<<<gemm_blocks(kN, kHD), TPB, 0, s>>>(
      w.xb, w.gWp + (size_t)li * kD * kHD, nullptr, w.f, kN, kD, kHD);

  k_el_er<<<gb_(NH * 32), TPB, 0, s>>>(w.f, p.attn_l + (size_t)li * kH * kD,
                                       p.attn_r + (size_t)li * kH * kD, w.el, w.er);
  k_fill_u32<<<gb_(NH), TPB, 0, s>>>(w.m1u, ENC_NEG_INF, NH);
  k_fill_f32<<<gb_(NH), TPB, 0, s>>>(w.s1, 0.f, NH);
  k_fill_f32<<<gb_(NHD), TPB, 0, s>>>(w.rst, 0.f, NHD);
  k_fill_u32<<<gb_(kN), TPB, 0, s>>>(w.m2u, ENC_NEG_INF, kN);
  k_fill_f32<<<gb_(kN), TPB, 0, s>>>(w.s2, 0.f, kN);
  k_fill_f32<<<gb_(ND), TPB, 0, s>>>(out, 0.f, ND);

  k_edge1_max<<<gb_(neH), TPB, 0, s>>>(w.el, w.er, src, dst, w.m1u, neH);
  k_edge1_exp<<<gb_(neH), TPB, 0, s>>>(w.el, w.er, src, dst, w.m1u, w.s1, w.ex1, neH);
  k_agg1<<<gb_(neHD), TPB, 0, s>>>(w.f, w.ex1, w.s1, src, dst, w.rst, neHD);

  k_hb<<<gb_(NHD), TPB, 0, s>>>(w.rst, p.gat_b + (size_t)li * kHD, w.hb, NHD);
  // z = hb @ fc_W[li] + fc_b[li]
  k_gemm_bf16<<<gemm_blocks(kN, kD), TPB, 0, s>>>(
      w.hb, w.fWp + (size_t)li * kHD * kD, p.fc_b + (size_t)li * kD, w.z, kN, kHD, kD);

  k_zlzr<<<gb_((long long)kN * 32), TPB, 0, s>>>(
      w.z, p.attn_fc_W + (size_t)li * 2 * kD, w.zl, w.zr);
  k_edge2_max<<<gb_(ne), TPB, 0, s>>>(w.zl, w.zr, src, dst, w.m2u, ne);
  k_edge2_exp<<<gb_(ne), TPB, 0, s>>>(w.zl, w.zr, src, dst, w.m2u, w.s2, w.ex2, ne);
  k_agg2<<<gb_(neD), TPB, 0, s>>>(w.z, w.ex2, w.s2, src, dst, out, neD);
}

static void pass2(hipStream_t s, const Scratch& w, const Wts& p,
                  const float* exer_table, const float* f2b,
                  const int* const* esrc, const int* const* edst, const int* nes,
                  const float* stuX, float* outStu) {
  hipMemcpyAsync(w.x2, exer_table, (size_t)kEXER * kD * 4, hipMemcpyDeviceToDevice, s);
  hipMemcpyAsync(w.x2 + (size_t)kEXER * kD, stuX, (size_t)kSTU * kD * 4,
                 hipMemcpyDeviceToDevice, s);
  k_cvt_bf16<<<gb_((long long)kN * kD), TPB, 0, s>>>(w.x2, w.xb, (long long)kN * kD);
  run_layer(s, w, p, 0, esrc[0], edst[0], nes[0], w.A);
  run_layer(s, w, p, 1, esrc[1], edst[1], nes[1], w.Bf);
  k_cat2<<<gb_((long long)kN * 256), TPB, 0, s>>>(w.A, w.Bf, w.catb, (long long)kN * 256);
  k_gemm_bf16<<<gemm_blocks(kSTU, kD), TPB, 0, s>>>(
      w.catb + (size_t)kEXER * 256, w.f2Wp, f2b, w.fus, kSTU, 256, kD);
  k_add<<<gb_((long long)kSTU * kD), TPB, 0, s>>>(stuX, w.fus, outStu,
                                                  (long long)kSTU * kD);
}

extern "C" void kernel_launch(void* const* d_in, const int* in_sizes, int n_in,
                              void* d_out_, int out_size, void* d_ws, size_t ws_size,
                              hipStream_t stream) {
  const float* stu_table = (const float*)d_in[0];
  const float* exer_table = (const float*)d_in[1];
  const float* gat_W = (const float*)d_in[2];
  Wts p;
  p.attn_l = (const float*)d_in[3];
  p.attn_r = (const float*)d_in[4];
  p.gat_b = (const float*)d_in[5];
  const float* fc_W = (const float*)d_in[6];
  p.fc_b = (const float*)d_in[7];
  p.attn_fc_W = (const float*)d_in[8];
  const float* f1W = (const float*)d_in[9];
  const float* f1b = (const float*)d_in[10];
  const float* f2W = (const float*)d_in[11];
  const float* f2b = (const float*)d_in[12];
  const int* esrc[5]; const int* edst[5]; int nes[5]; int maxNe = 0;
  for (int i = 0; i < 5; ++i) {
    esrc[i] = (const int*)d_in[13 + 2 * i];
    edst[i] = (const int*)d_in[14 + 2 * i];
    nes[i] = in_sizes[13 + 2 * i];
    if (nes[i] > maxNe) maxNe = nes[i];
  }
  const int* stu_id = (const int*)d_in[23];
  float* outp = (float*)d_out_;
  float* stu2 = outp;
  float* loss = outp + (size_t)kSTU * kD;

  // bump allocator on workspace
  char* wp = (char*)d_ws;
  auto alloc = [&](size_t bytes) -> char* {
    char* r = wp;
    wp += (bytes + 255) & ~(size_t)255;
    return r;
  };
  Scratch w;
  const size_t ND4 = (size_t)kN * kD * 4, SD4 = (size_t)kSTU * kD * 4;
  w.x0 = (float*)alloc(ND4);        w.x2 = (float*)alloc(ND4);
  w.A = (float*)alloc(ND4);         w.Bf = (float*)alloc(ND4);
  w.Cf = (float*)alloc(ND4);        w.Df = (float*)alloc(ND4);
  w.BP = (float*)alloc(ND4);
  w.stu1 = (float*)alloc(SD4);      w.stu1p = (float*)alloc(SD4);
  w.stu2p = (float*)alloc(SD4);
  w.f = (float*)alloc((size_t)kN * kHD * 4);
  w.rst = (float*)alloc((size_t)kN * kHD * 4);
  w.z = (float*)alloc(ND4);
  w.el = (float*)alloc((size_t)kN * kH * 4);
  w.er = (float*)alloc((size_t)kN * kH * 4);
  w.s1 = (float*)alloc((size_t)kN * kH * 4);
  w.m1u = (unsigned*)alloc((size_t)kN * kH * 4);
  w.zl = (float*)alloc((size_t)kN * 4);
  w.zr = (float*)alloc((size_t)kN * 4);
  w.s2 = (float*)alloc((size_t)kN * 4);
  w.m2u = (unsigned*)alloc((size_t)kN * 4);
  w.ex1 = (float*)alloc((size_t)maxNe * kH * 4);
  w.ex2 = (float*)alloc((size_t)maxNe * 4);
  w.fus = (float*)alloc(SD4);
  w.bn = (float*)alloc((size_t)kBATCH * kD * 4);
  w.pn = (float*)alloc((size_t)kBATCH * kD * 4);
  w.S = (float*)alloc(kD * 4);
  w.xb = (bf16_t*)alloc((size_t)kN * kD * 2);
  w.hb = (bf16_t*)alloc((size_t)kN * kHD * 2);
  w.catb = (bf16_t*)alloc((size_t)kN * 512 * 2);
  w.gWp = (bf16_t*)alloc((size_t)5 * kD * kHD * 2);
  w.fWp = (bf16_t*)alloc((size_t)5 * kHD * kD * 2);
  w.f1Wp = (bf16_t*)alloc((size_t)512 * kD * 2);
  w.f2Wp = (bf16_t*)alloc((size_t)256 * kD * 2);

  hipStream_t s = stream;

  // weight conversion + fragment-order packing (per layer)
  for (int li = 0; li < 5; ++li) {
    k_packW<<<gb_((long long)kD * kHD), TPB, 0, s>>>(
        gat_W + (size_t)li * kD * kHD, w.gWp + (size_t)li * kD * kHD,
        kD, kHD, (long long)kD * kHD);
    k_packW<<<gb_((long long)kHD * kD), TPB, 0, s>>>(
        fc_W + (size_t)li * kHD * kD, w.fWp + (size_t)li * kHD * kD,
        kHD, kD, (long long)kHD * kD);
  }
  k_packW<<<gb_(512LL * kD), TPB, 0, s>>>(f1W, w.f1Wp, 512, kD, 512LL * kD);
  k_packW<<<gb_(256LL * kD), TPB, 0, s>>>(f2W, w.f2Wp, 256, kD, 256LL * kD);

  // x = concat(exer, stu)
  hipMemcpyAsync(w.x0, exer_table, (size_t)kEXER * kD * 4, hipMemcpyDeviceToDevice, s);
  hipMemcpyAsync(w.x0 + (size_t)kEXER * kD, stu_table, SD4, hipMemcpyDeviceToDevice, s);
  k_cvt_bf16<<<gb_((long long)kN * kD), TPB, 0, s>>>(w.x0, w.xb, (long long)kN * kD);

  // ---- pass 1: five GAT layers on shared x ----
  run_layer(s, w, p, 0, esrc[0], edst[0], nes[0], w.A);
  run_layer(s, w, p, 1, esrc[1], edst[1], nes[1], w.Bf);
  run_layer(s, w, p, 2, esrc[2], edst[2], nes[2], w.Cf);
  run_layer(s, w, p, 3, esrc[3], edst[3], nes[3], w.Df);
  run_layer(s, w, p, 4, esrc[4], edst[4], nes[4], w.BP);

  // fus = cat(a,b,c,d)@f1W + f1b (student rows only); stu1 = stu + fus
  k_cat4<<<gb_((long long)kN * 512), TPB, 0, s>>>(w.A, w.Bf, w.Cf, w.Df, w.catb,
                                                  (long long)kN * 512);
  k_gemm_bf16<<<gemm_blocks(kSTU, kD), TPB, 0, s>>>(
      w.catb + (size_t)kEXER * 512, w.f1Wp, f1b, w.fus, kSTU, 512, kD);
  k_add<<<gb_((long long)kSTU * kD), TPB, 0, s>>>(stu_table, w.fus, w.stu1,
                                                  (long long)kSTU * kD);
  // fus_p = cat(a,bp,c,d)@f1W + f1b ; stu1_p = stu + fus_p
  k_cat4<<<gb_((long long)kN * 512), TPB, 0, s>>>(w.A, w.BP, w.Cf, w.Df, w.catb,
                                                  (long long)kN * 512);
  k_gemm_bf16<<<gemm_blocks(kSTU, kD), TPB, 0, s>>>(
      w.catb + (size_t)kEXER * 512, w.f1Wp, f1b, w.fus, kSTU, 512, kD);
  k_add<<<gb_((long long)kSTU * kD), TPB, 0, s>>>(stu_table, w.fus, w.stu1p,
                                                  (long long)kSTU * kD);

  // ---- pass 2: expr2 on both branches ----
  pass2(s, w, p, exer_table, f2b, esrc, edst, nes, w.stu1, stu2);     // -> d_out
  pass2(s, w, p, exer_table, f2b, esrc, edst, nes, w.stu1p, w.stu2p); // -> ws

  // ---- SSL loss ----
  k_gather_norm<<<gb_((long long)kBATCH * 32), TPB, 0, s>>>(stu2, stu_id, w.bn, kBATCH);
  k_gather_norm<<<gb_((long long)kBATCH * 32), TPB, 0, s>>>(w.stu2p, stu_id, w.pn, kBATCH);
  k_fill_f32<<<1, TPB, 0, s>>>(w.S, 0.f, kD);
  k_colsum<<<gb_((long long)kBATCH * kD), TPB, 0, s>>>(w.pn, w.S, (long long)kBATCH * kD);
  k_fill_f32<<<1, 32, 0, s>>>(loss, 0.f, 1);
  k_loss<<<gb_((long long)kBATCH * 32), TPB, 0, s>>>(w.bn, w.pn, w.S, loss);
}
